// ScaledDotProductAttention_28114855920344
// MI455X (gfx1250) — matrix operand order repaired and checked by reading, not compiled, not run on hardware
//
#include <hip/hip_runtime.h>
#include <hip/hip_bf16.h>

// Problem constants (match reference)
#define B_  8
#define S_  2048
#define DM  1024
#define DK  128

typedef __bf16 bf16;
typedef __attribute__((ext_vector_type(16))) __bf16 v16bf;
typedef __attribute__((ext_vector_type(8)))  float  v8f;

static __device__ __forceinline__ v8f wmma_bf16(v16bf a, v16bf b, v8f c) {
  // D = A(16x32 bf16) * B(32x16 bf16) + C(16x16 f32)
  return __builtin_amdgcn_wmma_f32_16x16x32_bf16(
      /*neg_a=*/false, a, /*neg_b=*/false, b,
      /*c_mod=*/(short)0, c, /*reuse_a=*/false, /*reuse_b=*/false);
}

// Async 16-byte global->LDS copy (CDNA5 Tensor/async path, ASYNCcnt-tracked).
// LDS offset = low 32 bits of the generic shared-aperture address (ISA 10.2).
static __device__ __forceinline__ void async_ld16(const bf16* g, bf16* l) {
  unsigned int       loff = (unsigned int)(unsigned long long)l;
  unsigned long long ga   = (unsigned long long)g;
  asm volatile("global_load_async_to_lds_b128 %0, %1, off"
               :: "v"(loff), "v"(ga) : "memory");
}
static __device__ __forceinline__ void wait_async0() {
  asm volatile("s_wait_asynccnt 0" ::: "memory");
}

// ---------------------------------------------------------------------------
// Kernel 0: transpose weights to bf16: WT[mat][n][k] = (bf16)W[mat][k][n]
//   3*DM*DK elements; tiny, one-shot, L2-resident thereafter.
// ---------------------------------------------------------------------------
__global__ __launch_bounds__(256)
void wt_kernel(const float* __restrict__ Wq, const float* __restrict__ Wk,
               const float* __restrict__ Wv, bf16* __restrict__ WT) {
  const int idx = blockIdx.x * 256 + threadIdx.x;  // [0, 3*DM*DK)
  const int mat = idx / (DM * DK);
  const int rem = idx - mat * (DM * DK);
  const int k   = rem / DK;
  const int n   = rem - k * DK;
  const float* W = (mat == 0) ? Wq : ((mat == 1) ? Wk : Wv);
  WT[(size_t)mat * DK * DM + (size_t)n * DM + k] = (bf16)W[(size_t)k * DK + n];
}

// ---------------------------------------------------------------------------
// Kernel 1: Q/K/V projection.  One wave computes one 16(M) x 16(N) tile,
// reducing over d_model=1024 in 32-wide K steps via v_wmma_f32_16x16x32_bf16.
// Q,K stored row-major [B*S][128] bf16; V stored TRANSPOSED [B][128][S] bf16
// so attention's P*V B-fragments are contiguous.
//   grid = (B*S/16, 6), block = 128 (4 waves); task in [0,24):
//   mat = task/8 (Q,K,V), ntile = task%8.
// ---------------------------------------------------------------------------
__global__ __launch_bounds__(128)
void qkv_proj_kernel(const float* __restrict__ x,
                     const bf16* __restrict__ WT,
                     bf16* __restrict__ Qb,
                     bf16* __restrict__ Kb,
                     bf16* __restrict__ VbT) {
  const int lane   = threadIdx.x & 31;
  const int wave   = threadIdx.x >> 5;
  const int task   = blockIdx.y * 4 + wave;   // 0..23
  const int mat    = task >> 3;               // 0=Q 1=K 2=V
  const int nt     = task & 7;                // column tile of d_k
  const int m0     = blockIdx.x * 16;         // row tile in [0, B*S)
  const int hi     = lane >> 4;               // lane half
  const int l16    = lane & 15;
  const int koffA  = hi * 8;                  // A-layout: interleaved K halves
  const int kbaseB = hi * 8;                  // B-layout: interleaved K halves

  const int    row  = m0 + l16;               // A row for this lane
  const float* xr   = x + (size_t)row * DM;
  const int    ncol = nt * 16 + l16;          // B/N column for this lane
  const bf16*  wt   = WT + ((size_t)mat * DK + ncol) * DM;  // WT row (contig k)

  v8f c;
#pragma unroll
  for (int r = 0; r < 8; ++r) c[r] = 0.0f;

  for (int k0 = 0; k0 < DM; k0 += 32) {
    v16bf a, b;
#pragma unroll
    for (int j = 0; j < 8; ++j) {
      a[j]     = (bf16)xr[k0 + koffA + j];            // K = koffA + j
      a[8 + j] = (bf16)xr[k0 + 16 + koffA + j];       // K = 16 + koffA + j
    }
#pragma unroll
    for (int e = 0; e < 16; ++e)                      // interleaved K halves
      b[e] = wt[k0 + kbaseB + e + (e & 8)];
    c = wmma_bf16(a, b, c);
  }

  // C/D layout: lane half selects M offset 0/8; VGPR r = M row within half.
  if (mat == 2) {
#pragma unroll
    for (int r = 0; r < 8; ++r) {
      const int grow = m0 + r + hi * 8;        // global row = b*S + s
      const int bb = grow >> 11;               // / 2048
      const int ss = grow & (S_ - 1);
      VbT[((size_t)bb * DK + ncol) * S_ + ss] = (bf16)c[r];
    }
  } else {
    bf16* Out = (mat == 0) ? Qb : Kb;
#pragma unroll
    for (int r = 0; r < 8; ++r)
      Out[(size_t)(m0 + r + hi * 8) * DK + ncol] = (bf16)c[r];
  }
}

// ---------------------------------------------------------------------------
// Kernel 2: flash attention.  Block = 4 waves (128 threads); each wave owns a
// 16-row query tile of one batch; K tile (32x128) and VT tile (128x32) are
// staged in LDS via async global->LDS loads and shared by all 4 waves.
//   grid = (S/64, B), block = 128.
// ---------------------------------------------------------------------------
__global__ __launch_bounds__(128)
void attn_kernel(const bf16* __restrict__ Qb,
                 const bf16* __restrict__ Kb,
                 const bf16* __restrict__ VbT,
                 float* __restrict__ out) {
  __shared__ __align__(16) bf16 Klds[32 * DK];     // [key][d]   8 KB
  __shared__ __align__(16) bf16 VldsT[DK * 32];    // [vdim][key] 8 KB
  __shared__ __align__(16) bf16 Plds[4][16 * 32];  // per-wave P scratch 4 KB

  const int lane   = threadIdx.x & 31;
  const int wave   = threadIdx.x >> 5;
  const int b      = blockIdx.y;
  const int q0     = (blockIdx.x * 4 + wave) * 16;
  const int hi     = lane >> 4;
  const int l16    = lane & 15;
  const int koffA  = hi * 8;
  const int kbaseB = hi * 8;

  // --- load the wave's Q tile (16 x 128) as 4 A-fragments, kept in VGPRs ---
  v16bf aq[4];
  {
    const bf16* qr = Qb + (size_t)(b * S_ + q0 + l16) * DK;
#pragma unroll
    for (int kb = 0; kb < 4; ++kb) {
#pragma unroll
      for (int j = 0; j < 8; ++j) {
        aq[kb][j]     = qr[kb * 32 + koffA + j];
        aq[kb][8 + j] = qr[kb * 32 + 16 + koffA + j];
      }
    }
  }

  v8f o[8];                     // output accumulators: 16 rows x 128 cols
#pragma unroll
  for (int n = 0; n < 8; ++n)
#pragma unroll
    for (int r = 0; r < 8; ++r) o[n][r] = 0.0f;

  float mrow[8], lrow[8];       // softmax stats for rows r + 8*hi
#pragma unroll
  for (int r = 0; r < 8; ++r) { mrow[r] = -1e30f; lrow[r] = 0.0f; }

  const float sc       = 0.08838834764831845f;  // 1/sqrt(128)
  const bf16* Kbase    = Kb  + (size_t)b * S_ * DK;   // [key][d]
  const bf16* VTbase   = VbT + (size_t)b * DK * S_;   // [vdim][key]

  for (int kt = 0; kt < S_ / 32; ++kt) {
    const int key0 = kt * 32;

    __syncthreads();  // all waves done reading previous tiles
    {
      // K tile: 32 keys x 128 d = one contiguous 8KB block
      // VT tile: 128 vdims x 32 keys (64B rows, stride S_)
#pragma unroll
      for (int i = 0; i < 4; ++i) {
        const int idx  = threadIdx.x + i * 128;       // 0..511 (16B units)
        async_ld16(Kbase + (size_t)key0 * DK + idx * 8, &Klds[idx * 8]);
        const int vdim = idx >> 2;
        const int part = idx & 3;
        async_ld16(VTbase + (size_t)vdim * S_ + key0 + part * 8,
                   &VldsT[idx * 8]);
      }
    }
    if (kt + 1 < S_ / 32) {  // pull next K tile toward L2 (global_prefetch_b8)
      __builtin_prefetch(Kbase + (size_t)(key0 + 32) * DK + threadIdx.x * 32, 0, 0);
    }
    wait_async0();    // this wave's async LDS writes complete
    __syncthreads();  // all waves' tiles staged

    // --- scores: S = Q(16x128) * K_tile^T(128x32) as two 16x16 frags ---
    v8f s0, s1;
#pragma unroll
    for (int r = 0; r < 8; ++r) { s0[r] = 0.0f; s1[r] = 0.0f; }
#pragma unroll
    for (int kb = 0; kb < 4; ++kb) {
      v16bf bk0, bk1;
      const bf16* k0p = &Klds[(size_t)l16 * DK + kb * 32 + kbaseB];        // keys 0..15
      const bf16* k1p = &Klds[(size_t)(16 + l16) * DK + kb * 32 + kbaseB]; // keys 16..31
#pragma unroll
      for (int e = 0; e < 16; ++e) { bk0[e] = k0p[e + (e & 8)]; bk1[e] = k1p[e + (e & 8)]; }
      s0 = wmma_bf16(aq[kb], bk0, s0);
      s1 = wmma_bf16(aq[kb], bk1, s1);
    }

    // --- online softmax over the 32 new keys (row = r + 8*hi) ---
#pragma unroll
    for (int r = 0; r < 8; ++r) {
      float v0 = s0[r] * sc;
      float v1 = s1[r] * sc;
      float mx = fmaxf(v0, v1);
#pragma unroll
      for (int msk = 1; msk < 16; msk <<= 1)       // reduce within 16-lane half
        mx = fmaxf(mx, __shfl_xor(mx, msk, 32));
      const float mnew = fmaxf(mrow[r], mx);
      const float p0   = __expf(v0 - mnew);
      const float p1   = __expf(v1 - mnew);
      float sum = p0 + p1;
#pragma unroll
      for (int msk = 1; msk < 16; msk <<= 1)
        sum += __shfl_xor(sum, msk, 32);
      const float alpha = __expf(mrow[r] - mnew);
      lrow[r] = lrow[r] * alpha + sum;
      mrow[r] = mnew;
#pragma unroll
      for (int n = 0; n < 8; ++n) o[n][r] *= alpha;  // rescale accumulators
      // stage P (bf16) for C->A re-layout; per-wave private LDS region
      Plds[wave][(r + 8 * hi) * 32 + l16]      = (bf16)p0;
      Plds[wave][(r + 8 * hi) * 32 + 16 + l16] = (bf16)p1;
    }
    __syncthreads();  // make P stores visible (ds wait)

    // --- O += P(16x32) * V_tile(32x128) ---
    v16bf ap;
    {
      const bf16* pp = &Plds[wave][(size_t)l16 * 32];
#pragma unroll
      for (int j = 0; j < 8; ++j) {
        ap[j]     = pp[koffA + j];
        ap[8 + j] = pp[16 + koffA + j];
      }
    }
#pragma unroll
    for (int nt = 0; nt < 8; ++nt) {
      v16bf bv;
      // VT layout: B[k=key][n=vdim] -> VldsT[vdim][key]; two 16B runs at keys 8*hi, 16+8*hi
      const bf16* vp = &VldsT[(size_t)(nt * 16 + l16) * 32 + kbaseB];
#pragma unroll
      for (int e = 0; e < 16; ++e) bv[e] = vp[e + (e & 8)];
      o[nt] = wmma_bf16(ap, bv, o[nt]);
    }
  }

  // --- normalize and store fp32 output [B,S,128] ---
#pragma unroll
  for (int r = 0; r < 8; ++r) {
    const float inv = 1.0f / lrow[r];
    float* orow = out + (size_t)(b * S_ + q0 + r + 8 * hi) * DK;
#pragma unroll
    for (int nt = 0; nt < 8; ++nt)
      orow[nt * 16 + l16] = o[nt][r] * inv;
  }
}

// ---------------------------------------------------------------------------
extern "C" void kernel_launch(void* const* d_in, const int* in_sizes, int n_in,
                              void* d_out, int out_size, void* d_ws, size_t ws_size,
                              hipStream_t stream) {
  (void)in_sizes; (void)n_in; (void)out_size; (void)ws_size;
  const float* x  = (const float*)d_in[0];
  // d_in[1] = enc_embed: unused (use_enc_embed=False path)
  const float* Wq = (const float*)d_in[2];
  const float* Wk = (const float*)d_in[3];
  const float* Wv = (const float*)d_in[4];
  float* out = (float*)d_out;

  // workspace: Q (4MB) + K (4MB) + V^T (4MB) bf16, + WT bf16 (768KB)
  bf16* Qb  = (bf16*)d_ws;
  bf16* Kb  = Qb  + (size_t)B_ * S_ * DK;
  bf16* VbT = Kb  + (size_t)B_ * S_ * DK;
  bf16* WT  = VbT + (size_t)B_ * S_ * DK;

  // 0) transpose + bf16-convert the three weight matrices
  wt_kernel<<<dim3((3 * DM * DK) / 256), 256, 0, stream>>>(Wq, Wk, Wv, WT);

  // 1) QKV projections: (B*S/16) row tiles x 24 wave-tasks (3 mats x 8 ntiles)
  qkv_proj_kernel<<<dim3((B_ * S_) / 16, 6), 128, 0, stream>>>(
      x, WT, Qb, Kb, VbT);

  // 2) flash attention: 4 query tiles (of 16 rows) per block, per batch
  attn_kernel<<<dim3(S_ / 64, B_), 128, 0, stream>>>(Qb, Kb, VbT, out);
}
